// LegalGAT_88708254532079
// MI455X (gfx1250) — matrix operand order repaired and checked by reading, not compiled, not run on hardware
//
#include <hip/hip_runtime.h>

// ---------------- problem constants (from the reference) ----------------
#define IN_C    1024
#define HEADS1  4
#define C1      64
#define HC1     256      // HEADS1*C1
#define OUT_C   64
#define NEG_SLOPE 0.2f

// ---------------- WMMA types ----------------
typedef __attribute__((ext_vector_type(16))) __bf16 v16bf;
typedef __attribute__((ext_vector_type(8)))  float  v8f;
typedef __attribute__((ext_vector_type(2)))  __bf16 bf16x2;
union Frag { v16bf v; unsigned u[8]; };

// native f32->bf16 (RNE) pair pack -> one dword (v_cvt_pk_bf16_f32 class op)
__device__ __forceinline__ unsigned packbf(float lo, float hi) {
  bf16x2 p;
  p.x = (__bf16)lo;
  p.y = (__bf16)hi;
  return __builtin_bit_cast(unsigned, p);
}

// monotone f32 <-> u32 encoding so atomicMax(u32) gives float max (incl. negatives)
__device__ __forceinline__ unsigned f2ord(float f) {
  unsigned u = __float_as_uint(f);
  return (u & 0x80000000u) ? ~u : (u | 0x80000000u);
}
__device__ __forceinline__ float ord2f(unsigned s) {
  unsigned u = (s & 0x80000000u) ? (s & 0x7FFFFFFFu) : ~s;
  return __uint_as_float(u);
}

// ---------------- bf16 WMMA GEMM:  C[M,N] = A[M,K] * W[N,K]^T ----------------
// block = 128 threads (4 waves), tile 64x64, K step 32
#define BM 64
#define BN 64
#define BK 32
#define LDK 40   // ushort row stride (32 data + 8 pad) => 80B rows, dword aligned

__global__ __launch_bounds__(128)
void gemm_bf16_wmma(const float* __restrict__ A, const float* __restrict__ W,
                    float* __restrict__ C, int M, int N, int K) {
  __shared__ unsigned short As[BM * LDK];   // [m][k]
  __shared__ unsigned short Bs[BN * LDK];   // [n][k]  (n-major so frag loads are dwords)

  const int tid  = threadIdx.x;
  const int wave = tid >> 5;
  const int lane = tid & 31;
  const int half = lane >> 4;
  const int l16  = lane & 15;
  const int m0   = blockIdx.x * BM;
  const int n0   = blockIdx.y * BN;

  v8f acc[4] = {};

  const int row = tid >> 1;          // 0..63 (tile row staged by this thread)
  const int kk  = (tid & 1) * 16;    // 0 or 16 within the 32-wide K slice

  for (int k0 = 0; k0 < K; k0 += BK) {
    // ---- stage A (f32 -> bf16 via native cvt, dword stores) ----
    {
      const int gm = m0 + row;
      float t[16];
      if (gm < M) {
        const float4* p = (const float4*)(A + (size_t)gm * K + k0 + kk);
        *(float4*)&t[0]  = p[0]; *(float4*)&t[4]  = p[1];
        *(float4*)&t[8]  = p[2]; *(float4*)&t[12] = p[3];
        if (k0 + BK < K) __builtin_prefetch(A + (size_t)gm * K + k0 + BK + kk, 0, 1);
      } else {
        #pragma unroll
        for (int j = 0; j < 16; ++j) t[j] = 0.f;
      }
      unsigned* d = (unsigned*)&As[row * LDK + kk];
      #pragma unroll
      for (int j = 0; j < 8; ++j) d[j] = packbf(t[2 * j], t[2 * j + 1]);
    }
    // ---- stage B = W[n][k] (f32 -> bf16), n-major ----
    {
      const float4* p = (const float4*)(W + (size_t)(n0 + row) * K + k0 + kk);
      float t[16];
      *(float4*)&t[0]  = p[0]; *(float4*)&t[4]  = p[1];
      *(float4*)&t[8]  = p[2]; *(float4*)&t[12] = p[3];
      if (k0 + BK < K) __builtin_prefetch(W + (size_t)(n0 + row) * K + k0 + BK + kk, 0, 1);
      unsigned* d = (unsigned*)&Bs[row * LDK + kk];
      #pragma unroll
      for (int j = 0; j < 8; ++j) d[j] = packbf(t[2 * j], t[2 * j + 1]);
    }
    __syncthreads();

    // ---- A fragment: 16x32 bf16, rows m = wave*16 + l16 ----
    Frag a;
    const unsigned* arow = (const unsigned*)&As[(wave * 16 + l16) * LDK];
    #pragma unroll
    for (int i = 0; i < 8; ++i) {
      const int dw = ((i < 4) ? 0 : 8) + half * 4 + (i & 3);  // ISA 16-bit A layout
      a.u[i] = arow[dw];
    }
    // ---- 4 B fragments / WMMAs across the 64-wide N tile ----
    #pragma unroll
    for (int ns = 0; ns < 4; ++ns) {
      Frag b;
      const unsigned* brow = (const unsigned*)&Bs[(ns * 16 + l16) * LDK];
      #pragma unroll
      for (int i = 0; i < 8; ++i) b.u[i] = brow[((i < 4) ? 0 : 8) + half * 4 + (i & 3)];  // K pairs per dword
      acc[ns] = __builtin_amdgcn_wmma_f32_16x16x32_bf16(
          false, a.v, false, b.v, (short)0, acc[ns], false, false);
    }
    __syncthreads();
  }

  // ---- store D: VGPR r -> row r (lanes 0-15) / r+8 (lanes 16-31), col = l16 ----
  const int gm_base = m0 + wave * 16 + half * 8;
  #pragma unroll
  for (int ns = 0; ns < 4; ++ns) {
    const int gn = n0 + ns * 16 + l16;
    #pragma unroll
    for (int r = 0; r < 8; ++r) {
      const int gm = gm_base + r;
      if (gm < M) C[(size_t)gm * N + gn] = acc[ns][r];
    }
  }
}

// ---------------- per-(node,head) attention logits: one wave each ----------------
__global__ __launch_bounds__(256)
void alpha_kernel(const float* __restrict__ h, const float* __restrict__ a_src,
                  const float* __restrict__ a_dst, float* __restrict__ as,
                  float* __restrict__ ad, int total, int H, int Cc) {
  const int w    = (int)((blockIdx.x * (long)blockDim.x + threadIdx.x) >> 5);
  const int lane = threadIdx.x & 31;
  if (w >= total) return;
  const int hd = w % H;
  const float* hp = h + (size_t)w * Cc;
  float ss = 0.f, sd = 0.f;
  for (int j = lane; j < Cc; j += 32) {
    const float v = hp[j];
    ss += v * a_src[hd * Cc + j];
    sd += v * a_dst[hd * Cc + j];
  }
  #pragma unroll
  for (int off = 16; off; off >>= 1) {
    ss += __shfl_down(ss, off, 32);
    sd += __shfl_down(sd, off, 32);
  }
  if (lane == 0) { as[w] = ss; ad[w] = sd; }
}

// ---------------- pass 1: segment max over destinations ----------------
__global__ __launch_bounds__(256)
void edge_max_kernel(const long long* __restrict__ esrc, const long long* __restrict__ edst,
                     const float* __restrict__ as, const float* __restrict__ ad,
                     unsigned* __restrict__ m, int E, int Etot, int H) {
  const int i = blockIdx.x * blockDim.x + threadIdx.x;
  if (i >= Etot) return;
  const int s = (i < E) ? (int)esrc[i] : (i - E);  // self-loops appended
  const int d = (i < E) ? (int)edst[i] : (i - E);
  for (int h = 0; h < H; ++h) {
    float e = as[s * H + h] + ad[d * H + h];
    e = (e > 0.f) ? e : NEG_SLOPE * e;
    atomicMax(&m[d * H + h], f2ord(e));
  }
}

// ---------------- pass 2: exp + denom + weighted scatter (wave per edge) ----------------
__global__ __launch_bounds__(256)
void edge_accum_kernel(const long long* __restrict__ esrc, const long long* __restrict__ edst,
                       const float* __restrict__ as, const float* __restrict__ ad,
                       const unsigned* __restrict__ m, const float* __restrict__ hfeat,
                       float* __restrict__ den, float* __restrict__ acc,
                       int E, int Etot, int H, int Cc) {
  const int w    = (int)((blockIdx.x * (long)blockDim.x + threadIdx.x) >> 5);
  const int lane = threadIdx.x & 31;
  if (w >= Etot) return;
  const int s  = (w < E) ? (int)esrc[w] : (w - E);
  const int d  = (w < E) ? (int)edst[w] : (w - E);
  const int HC = H * Cc;
  const int pl = HC >> 5;           // channels per lane (8 for layer1, 2 for layer2)
  const int ch0 = lane * pl;
  const int hd  = ch0 / Cc;         // this lane's head (constant across its channels)
  float e = as[s * H + hd] + ad[d * H + hd];
  e = (e > 0.f) ? e : NEG_SLOPE * e;
  const float ex = expf(e - ord2f(m[d * H + hd]));
  if ((ch0 % Cc) == 0) atomicAdd(&den[d * H + hd], ex);   // one lane per head
  const float* hs = hfeat + (size_t)s * HC;
  float*       ap = acc   + (size_t)d * HC;
  for (int j = 0; j < pl; ++j) {
    const int ch = ch0 + j;
    atomicAdd(&ap[ch], ex * hs[ch]);
  }
}

// ---------------- normalize + bias (+ELU) ----------------
__global__ __launch_bounds__(256)
void finalize_kernel(const float* __restrict__ acc_in, const float* __restrict__ den,
                     const float* __restrict__ bias, float* __restrict__ out,
                     long total, int H, int Cc, int elu) {
  const long i = blockIdx.x * (long)blockDim.x + threadIdx.x;
  if (i >= total) return;
  const int HC = H * Cc;
  const long n = i / HC;
  const int ch = (int)(i % HC);
  float v = acc_in[i] / den[n * H + ch / Cc] + bias[ch];
  if (elu) v = (v > 0.f) ? v : (expf(v) - 1.f);
  out[i] = v;
}

__global__ __launch_bounds__(256)
void zero_kernel(unsigned* __restrict__ p, long n) {
  const long i = blockIdx.x * (long)blockDim.x + threadIdx.x;
  if (i < n) p[i] = 0u;
}

// ---------------- host side ----------------
static inline int cdiv(long a, long b) { return (int)((a + b - 1) / b); }

extern "C" void kernel_launch(void* const* d_in, const int* in_sizes, int n_in,
                              void* d_out, int out_size, void* d_ws, size_t ws_size,
                              hipStream_t stream) {
  const float*     x      = (const float*)d_in[0];
  const long long* edge   = (const long long*)d_in[1];   // int64 [2,E]
  const float*     W1     = (const float*)d_in[2];       // [256,1024]
  const float*     a_src1 = (const float*)d_in[3];
  const float*     a_dst1 = (const float*)d_in[4];
  const float*     b1     = (const float*)d_in[5];
  const float*     W2     = (const float*)d_in[6];       // [64,256]
  const float*     a_src2 = (const float*)d_in[7];
  const float*     a_dst2 = (const float*)d_in[8];
  const float*     b2     = (const float*)d_in[9];

  const int N    = in_sizes[0] / IN_C;
  const int E    = in_sizes[1] / 2;
  const int Etot = E + N;                                // PyG add_self_loops
  const long long* esrc = edge;
  const long long* edst = edge + E;

  // carve workspace (all 256B aligned)
  char* wp = (char*)d_ws;
  auto carve = [&](size_t bytes) {
    void* p = (void*)wp;
    wp += (bytes + 255) & ~(size_t)255;
    return p;
  };
  float*    acc1 = (float*)   carve((size_t)N * HC1 * 4);
  float*    den1 = (float*)   carve((size_t)N * HEADS1 * 4);
  unsigned* m1   = (unsigned*)carve((size_t)N * HEADS1 * 4);
  float*    den2 = (float*)   carve((size_t)N * 4);
  unsigned* m2   = (unsigned*)carve((size_t)N * 4);
  float*    h1   = (float*)   carve((size_t)N * HC1 * 4);
  float*    as1  = (float*)   carve((size_t)N * HEADS1 * 4);
  float*    ad1  = (float*)   carve((size_t)N * HEADS1 * 4);
  float*    h2   = (float*)   carve((size_t)N * OUT_C * 4);
  float*    as2  = (float*)   carve((size_t)N * 4);
  float*    ad2  = (float*)   carve((size_t)N * 4);
  float*    dout = (float*)d_out;

  // ---- init accumulators (acc..m2 are contiguous in ws) + d_out ----
  {
    const long span = (long)(((char*)(m2 + N) - (char*)acc1) / 4);
    zero_kernel<<<cdiv(span, 256), 256, 0, stream>>>((unsigned*)acc1, span);
    zero_kernel<<<cdiv((long)N * OUT_C, 256), 256, 0, stream>>>((unsigned*)dout, (long)N * OUT_C);
  }

  // ================= layer 1 (H=4, C=64, concat) =================
  {
    dim3 g(cdiv(N, BM), HC1 / BN);
    gemm_bf16_wmma<<<g, 128, 0, stream>>>(x, W1, h1, N, HC1, IN_C);
  }
  alpha_kernel<<<cdiv((long)N * HEADS1, 8), 256, 0, stream>>>(h1, a_src1, a_dst1, as1, ad1,
                                                              N * HEADS1, HEADS1, C1);
  edge_max_kernel<<<cdiv(Etot, 256), 256, 0, stream>>>(esrc, edst, as1, ad1, m1, E, Etot, HEADS1);
  edge_accum_kernel<<<cdiv(Etot, 8), 256, 0, stream>>>(esrc, edst, as1, ad1, m1, h1,
                                                       den1, acc1, E, Etot, HEADS1, C1);
  // normalize + b1 + ELU, in place -> acc1 becomes layer-2 input [N,256]
  finalize_kernel<<<cdiv((long)N * HC1, 256), 256, 0, stream>>>(acc1, den1, b1, acc1,
                                                                (long)N * HC1, HEADS1, C1, 1);

  // ================= layer 2 (H=1, C=64, mean == identity) =================
  {
    dim3 g(cdiv(N, BM), OUT_C / BN);
    gemm_bf16_wmma<<<g, 128, 0, stream>>>(acc1, W2, h2, N, OUT_C, HC1);
  }
  alpha_kernel<<<cdiv((long)N, 8), 256, 0, stream>>>(h2, a_src2, a_dst2, as2, ad2, N, 1, OUT_C);
  edge_max_kernel<<<cdiv(Etot, 256), 256, 0, stream>>>(esrc, edst, as2, ad2, m2, E, Etot, 1);
  edge_accum_kernel<<<cdiv(Etot, 8), 256, 0, stream>>>(esrc, edst, as2, ad2, m2, h2,
                                                       den2, dout, E, Etot, 1, OUT_C);
  finalize_kernel<<<cdiv((long)N * OUT_C, 256), 256, 0, stream>>>(dout, den2, b2, dout,
                                                                  (long)N * OUT_C, 1, OUT_C, 0);
}
